// Prediction_3358664426280
// MI455X (gfx1250) — compile-verified, matrix operand order checked by reading, not run on hardware
//
#include <hip/hip_runtime.h>
#include <hip/hip_bf16.h>
#include <stdint.h>

// ---------------------------------------------------------------------------
// MI455X (gfx1250) implementation, round 2.
// All GEMMs on V_WMMA_F32_16X16X32_BF16 (wave32). Weights are pre-transposed
// and pre-converted to bf16 [n][k] in workspace (L2-resident, <= 13 MB), so
// the hot loops are pure global_load_b128 + wmma with no LDS staging/barriers.
// The big enc @ W_attn[H:] GEMM (~155 GFLOP) fuses the tanh()*v reduction so
// the [B,S1,H] (402 MB) intermediate never exists.
// ---------------------------------------------------------------------------

#define NEGV (-1e12f)

typedef __attribute__((ext_vector_type(16))) __bf16 bf16x16;
typedef __attribute__((ext_vector_type(8)))  __bf16 bf16x8;
typedef __attribute__((ext_vector_type(4)))  __bf16 bf16x4;
typedef __attribute__((ext_vector_type(8)))  float  f32x8;

#define WMMA_BF16(a, b, c) \
  __builtin_amdgcn_wmma_f32_16x16x32_bf16(false, (a), false, (b), (short)0, (c), false, false)

// CDNA5 16-bit A/B fragment lane layout (ISA 7.12.2):
// lanes 0-15: row = lane, K = {0..7} and {16..23}; lanes 16-31: row = lane-16,
// K = {8..15} and {24..31}. With row-major K-contiguous bf16 storage this is
// two contiguous 16B loads per lane (caller bakes row*ld + kchunk into p).
__device__ inline bf16x16 ldfrag(const __bf16* p) {
  union { bf16x16 v; bf16x8 h[2]; } u;
  u.h[0] = *(const bf16x8*)(p);
  u.h[1] = *(const bf16x8*)(p + 16);
  return u.v;
}

// Same fragment, but source is row-major f32: four float4 loads + pack to bf16.
__device__ inline bf16x16 ldfrag_f32(const float* q) {
  const float4 c0 = *(const float4*)(q);
  const float4 c1 = *(const float4*)(q + 4);
  const float4 c2 = *(const float4*)(q + 16);
  const float4 c3 = *(const float4*)(q + 20);
  union { bf16x16 v; __bf16 e[16]; } u;
  u.e[0]  = (__bf16)c0.x; u.e[1]  = (__bf16)c0.y; u.e[2]  = (__bf16)c0.z; u.e[3]  = (__bf16)c0.w;
  u.e[4]  = (__bf16)c1.x; u.e[5]  = (__bf16)c1.y; u.e[6]  = (__bf16)c1.z; u.e[7]  = (__bf16)c1.w;
  u.e[8]  = (__bf16)c2.x; u.e[9]  = (__bf16)c2.y; u.e[10] = (__bf16)c2.z; u.e[11] = (__bf16)c2.w;
  u.e[12] = (__bf16)c3.x; u.e[13] = (__bf16)c3.y; u.e[14] = (__bf16)c3.z; u.e[15] = (__bf16)c3.w;
  return u.v;
}

// ---------------------------------------------------------------------------
// Weight pre-pass: Wt[n*K + k] = (bf16)W[k*768 + n]   (one-time, L2-resident)
// Each thread produces 4 consecutive k's for one n -> one 8B store.
// ---------------------------------------------------------------------------
__global__ void wtrans_kernel(const float* __restrict__ W, __bf16* __restrict__ Wt, int K) {
  int idx = blockIdx.x * 256 + threadIdx.x;      // over (K/4)*768
  int n  = idx % 768;
  int k4 = (idx / 768) * 4;
  if (k4 >= K) return;
  bf16x4 o;
  o[0] = (__bf16)W[(size_t)(k4 + 0) * 768 + n];
  o[1] = (__bf16)W[(size_t)(k4 + 1) * 768 + n];
  o[2] = (__bf16)W[(size_t)(k4 + 2) * 768 + n];
  o[3] = (__bf16)W[(size_t)(k4 + 3) * 768 + n];
  *(bf16x4*)(Wt + (size_t)n * K + k4) = o;
}

// ---------------------------------------------------------------------------
// Generic GEMM: C[M,768] = A_bf16[M,K](lda) @ Bt_bf16[n][k](ldbt) + bias
// block = 256 (8 waves); tile 16 rows x 128 cols; inner loop: 4 loads + 1 wmma.
// ---------------------------------------------------------------------------
__global__ void gemm_bf16_kernel(const __bf16* __restrict__ A, int lda,
                                 const __bf16* __restrict__ Bt, int ldbt,
                                 const float* __restrict__ bias,
                                 float* __restrict__ C, int K) {
  const int tid = threadIdx.x, wave = tid >> 5, lane = tid & 31;
  const int m0 = blockIdx.x * 16;
  const int n0 = blockIdx.y * 128 + wave * 16;
  const int mr = lane & 15;
  const int kb = (lane < 16) ? 0 : 8;

  const __bf16* ap = A  + (size_t)(m0 + mr) * lda  + kb;
  const __bf16* bp = Bt + (size_t)(n0 + mr) * ldbt + kb;

  f32x8 acc = {};
  for (int k0 = 0; k0 < K; k0 += 32) {
    bf16x16 a = ldfrag(ap + k0);
    bf16x16 b = ldfrag(bp + k0);
    acc = WMMA_BF16(a, b, acc);
  }
  // C/D layout: VGPR j -> M = j + (lane<16 ? 0 : 8), N = lane&15
  const int n  = n0 + mr;
  const float bv = bias ? bias[n] : 0.f;
  const int mb = m0 + ((lane < 16) ? 0 : 8);
#pragma unroll
  for (int j = 0; j < 8; ++j)
    C[(size_t)(mb + j) * 768 + n] = acc[j] + bv;
}

// ---------------------------------------------------------------------------
// Fused row-score kernel:
//   out[row] = sum_n v[n] * tanh( (Arows[row,:768] @ W)[n] + extra[b,n] )
// Block tile: 32 rows x full N=768 (8 waves x 6 n-tiles x 2 m-tiles = 12 wmma
// per wave-step). Arows is f32, packed to bf16 in-register.
// ---------------------------------------------------------------------------
__global__ void fused_score_kernel(const float* __restrict__ Arows,
                                   const __bf16* __restrict__ Bt, int ldbt,
                                   const float* __restrict__ extra,  // [B,768]
                                   const float* __restrict__ v,      // [768]
                                   int rows_per_b,
                                   float* __restrict__ outp,
                                   const int* __restrict__ mask,     // nullptr or [B,out_stride]
                                   int out_stride, int out_col_off) {
  __shared__ float lds_e[32];
  const int tid = threadIdx.x, wave = tid >> 5, lane = tid & 31;
  const int r0 = blockIdx.x * 32;          // never crosses a batch group (64|512 % 32 == 0)
  const int b  = r0 / rows_per_b;
  const int mr = lane & 15;
  const int kb = (lane < 16) ? 0 : 8;

  const float*  ap0 = Arows + (size_t)(r0 + mr) * 768 + kb;
  const float*  ap1 = ap0 + (size_t)16 * 768;
  const __bf16* bp  = Bt + (size_t)(wave * 96 + mr) * ldbt + kb;

  f32x8 acc[2][6] = {};
  for (int k0 = 0; k0 < 768; k0 += 32) {
    __builtin_prefetch(bp + k0 + 32, 0, 1);          // next W k-slice
    bf16x16 a0 = ldfrag_f32(ap0 + k0);
    bf16x16 a1 = ldfrag_f32(ap1 + k0);
#pragma unroll
    for (int t = 0; t < 6; ++t) {
      bf16x16 bm = ldfrag(bp + (size_t)t * 16 * ldbt + k0);
      acc[0][t] = WMMA_BF16(a0, bm, acc[0][t]);
      acc[1][t] = WMMA_BF16(a1, bm, acc[1][t]);
    }
  }

  if (tid < 32) lds_e[tid] = 0.f;
  __syncthreads();

  const int joff = (lane < 16) ? 0 : 8;
#pragma unroll
  for (int mt = 0; mt < 2; ++mt) {
#pragma unroll
    for (int t = 0; t < 6; ++t) {
      const int n = wave * 96 + t * 16 + mr;
      const float ex = extra[(size_t)b * 768 + n];
      const float vn = v[n];
#pragma unroll
      for (int j = 0; j < 8; ++j) {
        float val = vn * tanhf(acc[mt][t][j] + ex);
        for (int s = 1; s < 16; s <<= 1) val += __shfl_xor(val, s, 32);  // reduce over n-lanes
        if (mr == 0) atomicAdd(&lds_e[mt * 16 + joff + j], val);
      }
    }
  }
  __syncthreads();
  if (tid < 32) {
    const int row = r0 + tid;
    const float e = lds_e[tid];
    if (mask) {
      const int bb = row / rows_per_b, s = row % rows_per_b;
      const int oi = bb * out_stride + out_col_off + s;
      outp[oi] = mask[oi] ? NEGV : e;
    } else {
      outp[row] = e;
    }
  }
}

// --------------------------- elementwise kernels ---------------------------

__global__ void convert_pre_kernel(const float* __restrict__ cur,
                                   const float* __restrict__ left,
                                   const float* __restrict__ opc,
                                   __bf16* __restrict__ cur_bf,
                                   __bf16* __restrict__ lc_bf,
                                   __bf16* __restrict__ opc_bf) {
  int i = blockIdx.x * 256 + threadIdx.x;            // 0 .. 393215
  if (i < 256 * 768) cur_bf[i] = (__bf16)cur[i];
  if (i < 64 * 768)  opc_bf[i] = (__bf16)opc[i];
  int b = i / 1536, h = i % 1536;
  lc_bf[i] = (__bf16)((h < 768) ? left[b * 768 + h] : cur[b * 768 + (h - 768)]);
}

__global__ void combine_node_kernel(const float* __restrict__ t1, const float* __restrict__ t2,
                                    const float* __restrict__ t3, const float* __restrict__ t4,
                                    const int* __restrict__ has_left,
                                    __bf16* __restrict__ leafcat,   // [B,1536], node half
                                    float* __restrict__ out_node) {
  int i = blockIdx.x * 256 + threadIdx.x;            // B*H
  int b = i / 768, h = i % 768;
  float gl = tanhf(t1[i]), tl = 1.f / (1.f + __expf(-t2[i]));
  float gr = tanhf(t3[i]), tr = 1.f / (1.f + __expf(-t4[i]));
  float nv = has_left[b] ? gr * tr : gl * tl;
  leafcat[(size_t)b * 1536 + h] = (__bf16)nv;
  out_node[i] = nv;
}

// one block per batch row: mask + softmax over S1=512 then context matvec
__global__ void softmax_context_kernel(const float* __restrict__ energy,
                                       const int* __restrict__ smask,
                                       const float* __restrict__ enc,
                                       float* __restrict__ out_ctx,
                                       __bf16* __restrict__ leafcat) {
  __shared__ float attn[512];
  __shared__ float red[8];
  const int b = blockIdx.x, tid = threadIdx.x;
  const int lane = tid & 31, wave = tid >> 5;
  const size_t base = (size_t)b * 512;

  float e0 = energy[base + tid];
  float e1 = energy[base + 256 + tid];
  if (smask[base + tid])       e0 = NEGV;
  if (smask[base + 256 + tid]) e1 = NEGV;

  float m = fmaxf(e0, e1);
  for (int s = 16; s > 0; s >>= 1) m = fmaxf(m, __shfl_xor(m, s, 32));
  if (lane == 0) red[wave] = m;
  __syncthreads();
  float gm = red[0];
  for (int i = 1; i < 8; ++i) gm = fmaxf(gm, red[i]);
  __syncthreads();

  float ex0 = __expf(e0 - gm), ex1 = __expf(e1 - gm);
  float s2 = ex0 + ex1;
  for (int s = 16; s > 0; s >>= 1) s2 += __shfl_xor(s2, s, 32);
  if (lane == 0) red[wave] = s2;
  __syncthreads();
  float gs = 0.f;
  for (int i = 0; i < 8; ++i) gs += red[i];
  float inv = 1.f / gs;
  attn[tid] = ex0 * inv;
  attn[tid + 256] = ex1 * inv;
  __syncthreads();

  for (int h = tid; h < 768; h += 256) {
    float c = 0.f;
    for (int s = 0; s < 512; ++s) c += attn[s] * enc[(base + s) * 768 + h];
    out_ctx[(size_t)b * 768 + h] = c;
    leafcat[(size_t)b * 1536 + 768 + h] = (__bf16)c;
  }
}

// scores for the batch-independent op/const candidates (o < 64)
__global__ void opscore_kernel(const float* __restrict__ leafpart,
                               const float* __restrict__ opW,
                               const float* __restrict__ v,
                               const int* __restrict__ cmask,
                               float* __restrict__ out) {
  __shared__ float red[8];
  const int b = blockIdx.x, o = blockIdx.y, tid = threadIdx.x;
  const int lane = tid & 31, wave = tid >> 5;
  float p = 0.f;
  for (int h = tid; h < 768; h += 256)
    p += v[h] * tanhf(leafpart[(size_t)b * 768 + h] + opW[(size_t)o * 768 + h]);
  for (int s = 16; s > 0; s >>= 1) p += __shfl_xor(p, s, 32);
  if (lane == 0) red[wave] = p;
  __syncthreads();
  if (tid == 0) {
    float sum = 0.f;
    for (int i = 0; i < 8; ++i) sum += red[i];
    out[b * 128 + o] = cmask[b * 128 + o] ? NEGV : sum;
  }
}

// emb_all output: [B,128,768] = concat(broadcast op_const, var_pades)
__global__ void embcopy_kernel(const float* __restrict__ opc,
                               const float* __restrict__ var,
                               float* __restrict__ out) {
  size_t i4 = (size_t)blockIdx.x * 256 + threadIdx.x;  // float4 units, 6291456 total
  size_t rem = i4;
  int h4 = (int)(rem % 192); rem /= 192;
  int o  = (int)(rem % 128);
  int b  = (int)(rem / 128);
  float4 val;
  if (o < 64) val = ((const float4*)opc)[(size_t)o * 192 + h4];
  else        val = ((const float4*)var)[((size_t)b * 64 + (o - 64)) * 192 + h4];
  ((float4*)out)[i4] = val;
}

// ---------------------------------------------------------------------------

extern "C" void kernel_launch(void* const* d_in, const int* in_sizes, int n_in,
                              void* d_out, int out_size, void* d_ws, size_t ws_size,
                              hipStream_t stream) {
  const float* cur     = (const float*)d_in[0];
  const float* left    = (const float*)d_in[1];
  const float* enc     = (const float*)d_in[2];
  const float* var     = (const float*)d_in[3];
  const float* opc     = (const float*)d_in[4];
  const float* W_l     = (const float*)d_in[5];
  const float* b_l     = (const float*)d_in[6];
  const float* W_lg    = (const float*)d_in[7];
  const float* b_lg    = (const float*)d_in[8];
  const float* W_r     = (const float*)d_in[9];
  const float* b_r     = (const float*)d_in[10];
  const float* W_rg    = (const float*)d_in[11];
  const float* b_rg    = (const float*)d_in[12];
  const float* W_attn  = (const float*)d_in[13];
  const float* b_attn  = (const float*)d_in[14];
  const float* v_attn  = (const float*)d_in[15];
  const float* W_score = (const float*)d_in[16];
  const float* b_score = (const float*)d_in[17];
  const float* v_score = (const float*)d_in[18];
  const int* has_left  = (const int*)d_in[19];
  const int* smask     = (const int*)d_in[20];
  const int* cmask     = (const int*)d_in[21];

  float* out = (float*)d_out;
  float* out_score = out;                 // [256,128]
  float* out_node  = out + 32768;         // [256,1,768]
  float* out_ctx   = out + 229376;        // [256,1,768]
  float* out_emb   = out + 425984;        // [256,128,768]

  char* ws = (char*)d_ws;
  __bf16* cur_bf   = (__bf16*)(ws + 0);          // 256*768 bf16
  __bf16* lc_bf    = (__bf16*)(ws + 393216);     // 256*1536 bf16
  __bf16* opc_bf   = (__bf16*)(ws + 1179648);    // 64*768 bf16
  float*  t1       = (float*)(ws + 1277952);     // 4x [256,768] f32
  float*  t2       = (float*)(ws + 2064384);
  float*  t3       = (float*)(ws + 2850816);
  float*  t4       = (float*)(ws + 3637248);
  __bf16* leafcat  = (__bf16*)(ws + 4423680);    // [256,1536] bf16 (node | context)
  float*  Aattn    = (float*)(ws + 5210112);     // [256,768] f32
  float*  energy   = (float*)(ws + 5996544);     // [256,512] f32
  float*  leafpart = (float*)(ws + 6520832);     // [256,768] f32
  float*  opW      = (float*)(ws + 7307264);     // [64,768] f32
  __bf16* Wt_l     = (__bf16*)(ws + 7503872);    // [768 n][768 k]
  __bf16* Wt_lg    = (__bf16*)(ws + 8683520);    // [768][768]
  __bf16* Wt_r     = (__bf16*)(ws + 9863168);    // [768][1536]
  __bf16* Wt_rg    = (__bf16*)(ws + 12222464);   // [768][1536]
  __bf16* Wt_attn  = (__bf16*)(ws + 14581760);   // [768][1536]
  __bf16* Wt_score = (__bf16*)(ws + 16941056);   // [768][2304]

  // 0) one-time weight transpose+convert to bf16 [n][k] (L2-resident)
  wtrans_kernel<<<576,  256, 0, stream>>>(W_l,     Wt_l,     768);
  wtrans_kernel<<<576,  256, 0, stream>>>(W_lg,    Wt_lg,    768);
  wtrans_kernel<<<1152, 256, 0, stream>>>(W_r,     Wt_r,     1536);
  wtrans_kernel<<<1152, 256, 0, stream>>>(W_rg,    Wt_rg,    1536);
  wtrans_kernel<<<1152, 256, 0, stream>>>(W_attn,  Wt_attn,  1536);
  wtrans_kernel<<<1728, 256, 0, stream>>>(W_score, Wt_score, 2304);

  // 1) f32 -> bf16 prep of small activations
  convert_pre_kernel<<<1536, 256, 0, stream>>>(cur, left, opc, cur_bf, lc_bf, opc_bf);

  // 2) four gate GEMMs
  gemm_bf16_kernel<<<dim3(16, 6), 256, 0, stream>>>(cur_bf, 768,  Wt_l,  768,  b_l,  t1, 768);
  gemm_bf16_kernel<<<dim3(16, 6), 256, 0, stream>>>(cur_bf, 768,  Wt_lg, 768,  b_lg, t2, 768);
  gemm_bf16_kernel<<<dim3(16, 6), 256, 0, stream>>>(lc_bf,  1536, Wt_r,  1536, b_r,  t3, 1536);
  gemm_bf16_kernel<<<dim3(16, 6), 256, 0, stream>>>(lc_bf,  1536, Wt_rg, 1536, b_rg, t4, 1536);

  // 3) node select; writes bf16 into leafcat[:, :768] and f32 node output
  combine_node_kernel<<<768, 256, 0, stream>>>(t1, t2, t3, t4, has_left, leafcat, out_node);

  // 4) A_attn = node @ W_attn[:H] + b_attn   (K=768 reads only k<768 of Wt_attn)
  gemm_bf16_kernel<<<dim3(16, 6), 256, 0, stream>>>(leafcat, 1536, Wt_attn, 1536, b_attn,
                                                    Aattn, 768);

  // 5) energy[b,s] = v_attn . tanh(A_attn[b] + enc[b,s] @ W_attn[H:])  -- big GEMM
  fused_score_kernel<<<4096, 256, 0, stream>>>(enc, Wt_attn + 768, 1536, Aattn, v_attn,
                                               512, energy, nullptr, 0, 0);

  // 6) mask + softmax + context; writes context half of leafcat
  softmax_context_kernel<<<256, 256, 0, stream>>>(energy, smask, enc, out_ctx, leafcat);

  // 7) leaf_part = [node|context] @ W_score[:2H] + b_score
  gemm_bf16_kernel<<<dim3(16, 6), 256, 0, stream>>>(leafcat, 1536, Wt_score, 2304, b_score,
                                                    leafpart, 1536);

  // 8) opW = emb_op_const @ W_score[2H:]  (batch-independent)
  gemm_bf16_kernel<<<dim3(4, 6), 256, 0, stream>>>(opc_bf, 768, Wt_score + 1536, 2304,
                                                   nullptr, opW, 768);

  // 9) var candidate scores (masked, into num_score[:,64:])
  fused_score_kernel<<<512, 256, 0, stream>>>(var, Wt_score + 1536, 2304, leafpart, v_score,
                                              64, out_score, cmask, 128, 64);

  // 10) op/const candidate scores (masked, into num_score[:,:64])
  opscore_kernel<<<dim3(256, 64), 256, 0, stream>>>(leafpart, opW, v_score, cmask, out_score);

  // 11) emb_all output copy
  embcopy_kernel<<<24576, 256, 0, stream>>>(opc, var, out_emb);
}
